// AutoRegGQAAttention_24034636988872
// MI455X (gfx1250) — matrix operand order repaired and checked by reading, not compiled, not run on hardware
//
#include <hip/hip_runtime.h>
#include <hip/hip_bf16.h>

#define B_  2
#define S_  2048
#define H_  2048
#define NH  16
#define HD  128
#define NKV 4

typedef __bf16 bf16_t;
typedef __attribute__((ext_vector_type(16))) __bf16 v16bf;
typedef __attribute__((ext_vector_type(8)))  __bf16 v8bf;
typedef __attribute__((ext_vector_type(8)))  float  v8f;

union V16U { v16bf v; struct { v8bf lo; v8bf hi; } p; };

// A-fragment (16x32 MxK, bf16): lane row = lane&15; lanes<16 hold K {0..7,16..23},
// lanes>=16 hold K {8..15,24..31}. base = row pointer already offset to k0.
static __device__ __forceinline__ v16bf load_afrag(const bf16_t* base, int hi) {
  V16U u;
  u.p.lo = *(const v8bf*)(base + hi * 8);
  u.p.hi = *(const v8bf*)(base + 16 + hi * 8);
  return u.v;
}
// B-fragment (32x16 KxN, bf16) from K-major rows: lane col = lane&15;
// lanes<16 hold K {0..7,16..23}, lanes>=16 hold K {8..15,24..31}.
static __device__ __forceinline__ v16bf load_bfrag(const bf16_t* base, int hi) {
  V16U u;
  const bf16_t* p = base + hi * 8;
  u.p.lo = *(const v8bf*)(p);
  u.p.hi = *(const v8bf*)(p + 16);
  return u.v;
}
static __device__ __forceinline__ v8f wmma_bf16(v16bf a, v16bf b, v8f c) {
  return __builtin_amdgcn_wmma_f32_16x16x32_bf16(false, a, false, b, (short)0, c,
                                                 false, false);
}

// ---------------- fp32 -> bf16 elementwise ----------------
__global__ void k_cvt_bf16(const float* __restrict__ in, bf16_t* __restrict__ out, int n) {
  int i = blockIdx.x * blockDim.x + threadIdx.x;
  if (i < n) out[i] = (bf16_t)in[i];
}

// ---------------- transpose + convert: W[K,N] fp32 -> out[N,K] bf16 ----------------
__global__ __launch_bounds__(256) void k_transpose_cvt(const float* __restrict__ W,
                                                       bf16_t* __restrict__ out,
                                                       int K, int N) {
  __shared__ float tile[32][33];
  const int k0 = blockIdx.y * 32, n0 = blockIdx.x * 32;
  const int tx = threadIdx.x & 31, ty = threadIdx.x >> 5;  // 32 x 8
#pragma unroll
  for (int i = 0; i < 32; i += 8)
    tile[ty + i][tx] = W[(size_t)(k0 + ty + i) * N + n0 + tx];
  __syncthreads();
#pragma unroll
  for (int i = 0; i < 32; i += 8)
    out[(size_t)(n0 + ty + i) * K + k0 + tx] = (bf16_t)tile[tx][ty + i];
}

// ---------------- GEMM: C[M,N] = A[M,K](bf16) * Bt[N,K](bf16)^T (+bias) ----------------
// 256 threads = 8 waves; each wave computes a 32x64 tile via v_wmma_f32_16x16x32_bf16.
__global__ __launch_bounds__(256) void k_gemm_bf16(const bf16_t* __restrict__ A,
                                                   const bf16_t* __restrict__ Bt,
                                                   float* __restrict__ C,
                                                   const float* __restrict__ bias,
                                                   int M, int N, int K, int ldc) {
  const int lane = threadIdx.x & 31, wave = threadIdx.x >> 5;
  const int hi = lane >> 4, ln = lane & 15;
  const int m0 = blockIdx.y * 256 + wave * 32;
  const int n0 = blockIdx.x * 64;

  const v8f zv = {0.f, 0.f, 0.f, 0.f, 0.f, 0.f, 0.f, 0.f};
  v8f acc[2][4];
#pragma unroll
  for (int mi = 0; mi < 2; mi++)
#pragma unroll
    for (int j = 0; j < 4; j++) acc[mi][j] = zv;

  const bf16_t* a0 = A + (size_t)(m0 + ln) * K;
  const bf16_t* a1 = A + (size_t)(m0 + 16 + ln) * K;
  const bf16_t* br[4];
#pragma unroll
  for (int j = 0; j < 4; j++) br[j] = Bt + (size_t)(n0 + j * 16 + ln) * K;

  for (int k0 = 0; k0 < K; k0 += 32) {
    v16bf af0 = load_afrag(a0 + k0, hi);
    v16bf af1 = load_afrag(a1 + k0, hi);
#pragma unroll
    for (int j = 0; j < 4; j++) {
      v16bf bfj = load_bfrag(br[j] + k0, hi);
      acc[0][j] = wmma_bf16(af0, bfj, acc[0][j]);
      acc[1][j] = wmma_bf16(af1, bfj, acc[1][j]);
    }
  }
#pragma unroll
  for (int mi = 0; mi < 2; mi++) {
#pragma unroll
    for (int j = 0; j < 4; j++) {
      const int n = n0 + j * 16 + ln;
      const float bb = bias ? bias[n] : 0.0f;
#pragma unroll
      for (int r = 0; r < 8; r++) {
        const int m = m0 + mi * 16 + r + 8 * hi;
        C[(size_t)m * ldc + n] = acc[mi][j][r] + bb;
      }
    }
  }
}

// ---------------- RoPE + bias + scatter ----------------
// qkv: [B*S, 3072] fp32 (Q | K | V). Writes:
//   Qb  [B,16,S,128]  bf16 (rope'd)
//   Kb  [B,4, S,128]  bf16 (rope'd)        outK [B,4,S,128] fp32 (new_past_key)
//   Vt  [B,4, 128,S]  bf16 (transposed)    outV [B,4,S,128] fp32 (new_past_value)
__global__ __launch_bounds__(256) void k_rope_scatter(
    const float* __restrict__ qkv, const float* __restrict__ bq,
    const float* __restrict__ bk, const float* __restrict__ bv,
    bf16_t* __restrict__ Qb, bf16_t* __restrict__ Kb, bf16_t* __restrict__ Vt,
    float* __restrict__ outK, float* __restrict__ outV) {
  const int s = blockIdx.x, b = blockIdx.y;
  const float* row = qkv + ((size_t)b * S_ + s) * 3072;
  const float kln = 9.210340371976184f / 64.0f;  // ln(10000)/ (d/2)

  // Q: 16 heads * 64 rope pairs
  for (int t = threadIdx.x; t < NH * 64; t += blockDim.x) {
    const int h = t >> 6, d2 = t & 63;
    const float ang = (float)s * __expf(-(float)d2 * kln);
    const float c = cosf(ang), sn = sinf(ang);
    const float x1 = row[h * HD + d2] + bq[h * HD + d2];
    const float x2 = row[h * HD + 64 + d2] + bq[h * HD + 64 + d2];
    const size_t o = (((size_t)b * NH + h) * S_ + s) * HD;
    Qb[o + d2]      = (bf16_t)(x1 * c - x2 * sn);
    Qb[o + 64 + d2] = (bf16_t)(x1 * sn + x2 * c);
  }
  // K: 4 heads * 64 rope pairs
  for (int t = threadIdx.x; t < NKV * 64; t += blockDim.x) {
    const int h = t >> 6, d2 = t & 63;
    const float ang = (float)s * __expf(-(float)d2 * kln);
    const float c = cosf(ang), sn = sinf(ang);
    const float x1 = row[2048 + h * HD + d2] + bk[h * HD + d2];
    const float x2 = row[2048 + h * HD + 64 + d2] + bk[h * HD + 64 + d2];
    const float r1 = x1 * c - x2 * sn, r2 = x1 * sn + x2 * c;
    const size_t o = (((size_t)b * NKV + h) * S_ + s) * HD;
    Kb[o + d2] = (bf16_t)r1;       Kb[o + 64 + d2] = (bf16_t)r2;
    outK[o + d2] = r1;             outK[o + 64 + d2] = r2;
  }
  // V: 4 heads * 128 dims
  for (int t = threadIdx.x; t < NKV * HD; t += blockDim.x) {
    const int h = t >> 7, d = t & 127;
    const float v = row[2560 + h * HD + d] + bv[h * HD + d];
    outV[(((size_t)b * NKV + h) * S_ + s) * HD + d] = v;
    Vt[(((size_t)b * NKV + h) * HD + d) * S_ + s] = (bf16_t)v;
  }
}

// ---------------- flash attention (causal, GQA) ----------------
// Q [B,16,S,128] bf16, Kb [B,4,S,128] bf16, Vt [B,4,128,S] bf16
// -> A2 [B*S, 2048] bf16 (attn output laid out as the next GEMM's A matrix)
__global__ __launch_bounds__(256) void k_attn(const bf16_t* __restrict__ Q,
                                              const bf16_t* __restrict__ Kb,
                                              const bf16_t* __restrict__ Vt,
                                              bf16_t* __restrict__ A2) {
  __shared__ bf16_t P[8][16 * 32];  // per-wave 16x32 probability tile
  const int lane = threadIdx.x & 31, wave = threadIdx.x >> 5;
  const int hi = lane >> 4, ln = lane & 15;
  const int b = blockIdx.z, h = blockIdx.y;
  const int q0 = blockIdx.x * 128 + wave * 16;  // this wave's 16 query rows
  const int hk = h >> 2;                        // GQA group

  const bf16_t* Qp = Q + (((size_t)b * NH + h) * S_ + q0) * HD;
  const bf16_t* Kp = Kb + (((size_t)b * NKV + hk) * S_) * HD;
  const bf16_t* Vp = Vt + (((size_t)b * NKV + hk) * HD) * S_;

  // Q A-fragments for the full head dim (4 chunks of K=32)
  v16bf qf[4];
  const bf16_t* qrow = Qp + (size_t)ln * HD;
#pragma unroll
  for (int kk = 0; kk < 4; kk++) qf[kk] = load_afrag(qrow + kk * 32, hi);

  const v8f zv = {0.f, 0.f, 0.f, 0.f, 0.f, 0.f, 0.f, 0.f};
  v8f o[8];
#pragma unroll
  for (int j = 0; j < 8; j++) o[j] = zv;
  float mrow[8], lrow[8];
#pragma unroll
  for (int r = 0; r < 8; r++) { mrow[r] = -1e30f; lrow[r] = 0.f; }

  const float scale = 0.08838834764831845f;  // 1/sqrt(128)
  const int qmax = q0 + 15;

  for (int k0 = 0; k0 <= qmax; k0 += 32) {
    // ---- scores: two 16x16 tiles over 32 keys ----
    v8f s0 = zv, s1 = zv;
#pragma unroll
    for (int kk = 0; kk < 4; kk++) {
      v16bf b0 = load_bfrag(Kp + (size_t)(k0 + ln) * HD + kk * 32, hi);
      v16bf b1 = load_bfrag(Kp + (size_t)(k0 + 16 + ln) * HD + kk * 32, hi);
      s0 = wmma_bf16(qf[kk], b0, s0);
      s1 = wmma_bf16(qf[kk], b1, s1);
    }
    // ---- online softmax update (per row r = r + 8*hi of this wave) ----
    float corr[8];
#pragma unroll
    for (int r = 0; r < 8; r++) {
      const int qi = q0 + r + 8 * hi;
      float a0 = s0[r] * scale;
      float a1 = s1[r] * scale;
      if (k0 + ln > qi) a0 = -1e30f;          // causal mask
      if (k0 + 16 + ln > qi) a1 = -1e30f;
      float mx = fmaxf(a0, a1);
#pragma unroll
      for (int msk = 1; msk < 16; msk <<= 1) mx = fmaxf(mx, __shfl_xor(mx, msk, 32));
      const float mnew = fmaxf(mrow[r], mx);
      const float c = __expf(mrow[r] - mnew);
      const float p0 = __expf(a0 - mnew);
      const float p1 = __expf(a1 - mnew);
      float sum = p0 + p1;
#pragma unroll
      for (int msk = 1; msk < 16; msk <<= 1) sum += __shfl_xor(sum, msk, 32);
      lrow[r] = lrow[r] * c + sum;
      mrow[r] = mnew;
      corr[r] = c;
      // stage P to LDS in C-layout; re-read below in A-layout (same-wave DS is in-order)
      P[wave][(r + 8 * hi) * 32 + ln] = (bf16_t)p0;
      P[wave][(r + 8 * hi) * 32 + 16 + ln] = (bf16_t)p1;
    }
    // rescale output accumulator
#pragma unroll
    for (int j = 0; j < 8; j++)
#pragma unroll
      for (int r = 0; r < 8; r++) o[j][r] *= corr[r];

    // ---- P x V : 8 WMMAs over the 128 head dims ----
    v16bf pf = load_afrag(&P[wave][ln * 32], hi);
#pragma unroll
    for (int j = 0; j < 8; j++) {
      v16bf vf = load_bfrag(Vp + (size_t)(j * 16 + ln) * S_ + k0, hi);
      o[j] = wmma_bf16(pf, vf, o[j]);
    }
  }

  // epilogue: normalize, convert, store as next GEMM's activation matrix
#pragma unroll
  for (int j = 0; j < 8; j++) {
#pragma unroll
    for (int r = 0; r < 8; r++) {
      const int row = q0 + r + 8 * hi;
      const float v = o[j][r] / lrow[r];
      A2[((size_t)b * S_ + row) * H_ + h * HD + j * 16 + ln] = (bf16_t)v;
    }
  }
}

extern "C" void kernel_launch(void* const* d_in, const int* in_sizes, int n_in,
                              void* d_out, int out_size, void* d_ws, size_t ws_size,
                              hipStream_t stream) {
  (void)in_sizes; (void)n_in; (void)out_size; (void)ws_size;
  const float* hidden = (const float*)d_in[0];
  // d_in[1] = mask (implicit causal; unused)
  const float* Wq = (const float*)d_in[2];
  const float* bq = (const float*)d_in[3];
  const float* Wk = (const float*)d_in[4];
  const float* bk = (const float*)d_in[5];
  const float* Wv = (const float*)d_in[6];
  const float* bv = (const float*)d_in[7];
  const float* Wo = (const float*)d_in[8];
  const float* bo = (const float*)d_in[9];
  float* out = (float*)d_out;

  char* ws = (char*)d_ws;
  size_t off = 0;
  auto wsalloc = [&](size_t bytes) {
    char* p = ws + off;
    off += (bytes + 255) & ~(size_t)255;
    return p;
  };
  bf16_t* Xb    = (bf16_t*)wsalloc((size_t)B_ * S_ * H_ * 2);       // X in bf16
  bf16_t* WqkvT = (bf16_t*)wsalloc((size_t)3072 * 2048 * 2);        // [Wq|Wk|Wv]^T bf16
  bf16_t* WoT   = (bf16_t*)wsalloc((size_t)2048 * 2048 * 2);        // Wo^T bf16
  float*  QKV   = (float*) wsalloc((size_t)B_ * S_ * 3072 * 4);     // raw QKV fp32
  bf16_t* Qb    = (bf16_t*)wsalloc((size_t)B_ * NH * S_ * HD * 2);  // rope'd Q
  bf16_t* Kbuf  = (bf16_t*)wsalloc((size_t)B_ * NKV * S_ * HD * 2); // rope'd K
  bf16_t* Vt    = (bf16_t*)wsalloc((size_t)B_ * NKV * HD * S_ * 2); // V transposed
  bf16_t* A2    = (bf16_t*)wsalloc((size_t)B_ * S_ * H_ * 2);       // attn out bf16

  float* outK = out + (size_t)B_ * S_ * H_;
  float* outV = outK + (size_t)B_ * NKV * S_ * HD;

  { // activations fp32 -> bf16
    int n = B_ * S_ * H_;
    k_cvt_bf16<<<(n + 255) / 256, 256, 0, stream>>>(hidden, Xb, n);
  }
  // weights -> transposed bf16 (B-fragments become contiguous K-runs)
  k_transpose_cvt<<<dim3(2048 / 32, 2048 / 32), 256, 0, stream>>>(Wq, WqkvT, 2048, 2048);
  k_transpose_cvt<<<dim3(512 / 32, 2048 / 32), 256, 0, stream>>>(Wk, WqkvT + (size_t)2048 * 2048, 2048, 512);
  k_transpose_cvt<<<dim3(512 / 32, 2048 / 32), 256, 0, stream>>>(Wv, WqkvT + (size_t)2560 * 2048, 2048, 512);
  k_transpose_cvt<<<dim3(2048 / 32, 2048 / 32), 256, 0, stream>>>(Wo, WoT, 2048, 2048);

  // fused QKV projection: [4096,2048] x [2048,3072]
  k_gemm_bf16<<<dim3(3072 / 64, 4096 / 256), 256, 0, stream>>>(
      Xb, WqkvT, QKV, nullptr, 4096, 3072, 2048, 3072);

  // RoPE + bias + layout scatter (also writes new_past_key/value outputs)
  k_rope_scatter<<<dim3(S_, B_), 256, 0, stream>>>(QKV, bq, bk, bv, Qb, Kbuf, Vt,
                                                   outK, outV);

  // causal flash attention
  k_attn<<<dim3(S_ / 128, NH, B_), 256, 0, stream>>>(Qb, Kbuf, Vt, A2);

  // output projection: [4096,2048] x [2048,2048] + bo
  k_gemm_bf16<<<dim3(2048 / 64, 4096 / 256), 256, 0, stream>>>(
      A2, WoT, out, bo, 4096, 2048, 2048, 2048);
}
